// VisionMambaEncoder_88338887344538
// MI455X (gfx1250) — compile-verified, matrix operand order checked by reading, not run on hardware
//
#include <hip/hip_runtime.h>
#include <cstdint>
#include <cstddef>

// ---------------- CDNA5 WMMA types ----------------
typedef __attribute__((ext_vector_type(16))) __bf16 bf16x16;
typedef __attribute__((ext_vector_type(8)))  float  f32x8;

union FragBF16 { bf16x16 v; uint32_t u[8]; };

// Types for the async global->LDS builtin (param 1 diagnosed as 'int __vector(4) *')
typedef int v4i __attribute__((vector_size(16)));
typedef __attribute__((address_space(3))) v4i v4i_as3;

#if defined(__has_builtin)
#if __has_builtin(__builtin_amdgcn_global_load_async_to_lds_b128)
#define HAS_ASYNC_LDS 1
#endif
#endif
#ifndef HAS_ASYNC_LDS
#define HAS_ASYNC_LDS 0
#endif

// Model constants
#define BATCH   8
#define IMG     128
#define HW      32          // 32x32 after patch embed
#define LSEQ    1024        // HW*HW
#define NPIX    (BATCH*LSEQ)   // 8192 token positions
#define DMODEL  128
#define DINNER  256
#define DSTATE  16
#define DTRANK  8
#define KDIR    4
#define CPAD    48          // 40 (dt+B+C) padded to 48 for 16-wide WMMA tiles

__device__ __forceinline__ uint16_t f2bf(float f) {
    uint32_t u = __float_as_uint(f);
    uint32_t r = u + 0x7FFFu + ((u >> 16) & 1u);   // round-to-nearest-even
    return (uint16_t)(r >> 16);
}
__device__ __forceinline__ float silu(float x) { return x / (1.f + __expf(-x)); }

// 16-byte global -> LDS transfer: async DMA (ASYNCcnt) when available.
__device__ __forceinline__ void g2l_16B(const void* g, void* l) {
#if HAS_ASYNC_LDS
    v4i*     gp = (v4i*)(uintptr_t)g;
    v4i_as3* lp = (v4i_as3*)(uint32_t)(uintptr_t)l;  // generic LDS addr low 32b == LDS offset
    __builtin_amdgcn_global_load_async_to_lds_b128(gp, lp, 0, 0);
#else
    *(uint4*)l = *(const uint4*)g;
#endif
}

template <int N>
__device__ __forceinline__ void wait_async() {
#if HAS_ASYNC_LDS
#if defined(__has_builtin) && __has_builtin(__builtin_amdgcn_s_wait_asynccnt)
    __builtin_amdgcn_s_wait_asynccnt(N);
#else
    asm volatile("s_wait_asynccnt %0" ::"i"(N) : "memory");
#endif
#endif
}

// ---------------- small prep kernels ----------------
__global__ void k_f32_to_bf16(const float* __restrict__ s, uint16_t* __restrict__ d, int n) {
    int i = blockIdx.x * blockDim.x + threadIdx.x;
    if (i < n) d[i] = f2bf(s[i]);
}

// x_proj_w (K,40,256) -> bf16 padded (K,48,256)
__global__ void k_xproj_pad(const float* __restrict__ s, uint16_t* __restrict__ d) {
    int i = blockIdx.x * blockDim.x + threadIdx.x;
    int n = KDIR * CPAD * DINNER;
    if (i >= n) return;
    int dd = i % DINNER;
    int c  = (i / DINNER) % CPAD;
    int k  = i / (DINNER * CPAD);
    float v = (c < 40) ? s[((size_t)k * 40 + c) * DINNER + dd] : 0.f;
    d[i] = f2bf(v);
}

__global__ void k_aneg(const float* __restrict__ alog, float* __restrict__ an, int n) {
    int i = blockIdx.x * blockDim.x + threadIdx.x;
    if (i < n) an[i] = -__expf(alog[i]);
}

// ---------------- patch embed (4x4 stride-4 conv, Cin=3) + BatchNorm -> bf16 tokens ----------------
__global__ void k_patch_embed(const float* __restrict__ x, const float* __restrict__ pw,
                              const float* __restrict__ gam, const float* __restrict__ bet,
                              const float* __restrict__ mean, const float* __restrict__ var,
                              uint16_t* __restrict__ xe) {
    int bp = blockIdx.x;
    int b = bp >> 10, pix = bp & 1023;
    int h = pix >> 5, w = pix & 31;
    int dm = threadIdx.x;
    float acc = 0.f;
#pragma unroll
    for (int ci = 0; ci < 3; ++ci)
#pragma unroll
        for (int ph = 0; ph < 4; ++ph)
#pragma unroll
            for (int pq = 0; pq < 4; ++pq) {
                float xv = x[(((size_t)b * 3 + ci) * IMG + (h * 4 + ph)) * IMG + (w * 4 + pq)];
                float wv = pw[(((size_t)dm * 3 + ci) * 4 + ph) * 4 + pq];
                acc += xv * wv;
            }
    float s = gam[dm] * rsqrtf(var[dm] + 1e-5f);
    xe[(size_t)bp * DMODEL + dm] = f2bf((acc - mean[dm]) * s + bet[dm]);
}

// ================= double-buffered async-LDS staged WMMA GEMM =================
// C[M,N] = A[M,K] * W[N,K]^T.  Block: 256 threads (8 waves), 128 x (NTILES*16) C tile.
// Per 32-deep K chunk: stage A(128x32) and B(BN x 32) bf16 in LDS via async DMA,
// double-buffered so WMMAs on chunk i overlap the DMA of chunk i+1.
template <int NTILES>
__global__ void k_gemm_staged(const uint16_t* __restrict__ A, const uint16_t* __restrict__ W,
                              float* __restrict__ C, int M, int N, int K) {
    constexpr int BN    = NTILES * 16;
    constexpr int SEG_A = 128 * 32 * 2 / 16;          // 16B segments per A chunk (512)
    constexpr int SEG_B = BN * 32 * 2 / 16;           // segments per B chunk (512 or 192)
    constexpr int PER_A = SEG_A / 256;                // 2
    constexpr int PER_B = (SEG_B + 255) / 256;        // 2 or 1 (padded to uniform)
    constexpr int PER   = PER_A + PER_B;              // per-thread async issues per chunk

    __shared__ uint16_t sA[2][128 * 32];
    __shared__ uint16_t sB[2][BN * 32];

    const int tid  = threadIdx.x;
    const int wv   = tid >> 5;
    const int lane = tid & 31;
    const int hf   = lane >> 4, r = lane & 15;
    const int m0   = blockIdx.x * 128;
    const int n0   = blockIdx.y * BN;

    f32x8 acc[NTILES];
#pragma unroll
    for (int t = 0; t < NTILES; ++t) acc[t] = {};

    const int nch = K >> 5;

    auto issue = [&](int ch) {
        const int buf = ch & 1;
#pragma unroll
        for (int p = 0; p < PER_A; ++p) {
            int s = tid + p * 256;                    // 0..511
            int row = s >> 2, part = s & 3;
            const uint16_t* g = A + (size_t)(m0 + row) * K + ch * 32 + part * 8;
            g2l_16B(g, &sA[buf][s * 8]);
        }
#pragma unroll
        for (int p = 0; p < PER_B; ++p) {
            int s = tid + p * 256;
            if (s >= SEG_B) s -= SEG_B;               // uniform issue count (benign dup write)
            int row = s >> 2, part = s & 3;
            const uint16_t* g = W + (size_t)(n0 + row) * K + ch * 32 + part * 8;
            g2l_16B(g, &sB[buf][s * 8]);
        }
    };

    issue(0);
    for (int ch = 0; ch < nch; ++ch) {
        if (ch + 1 < nch) {                           // overlap DMA of next chunk with compute
            issue(ch + 1);
            wait_async<PER>();                        // drain chunk ch only (in-order per wave)
        } else {
            wait_async<0>();
        }
        __syncthreads();

        const int buf = ch & 1;
        FragBF16 fa;
        const uint16_t* pa = &sA[buf][(wv * 16 + r) * 32];
#pragma unroll
        for (int v = 0; v < 8; ++v) {
            // ISA 16-bit A 16x32 layout: v<4 -> K = 2v + 8*half ; v>=4 -> K = 16 + 2(v-4) + 8*half
            int kb = ((v < 4) ? (2 * v) : (16 + 2 * (v - 4))) + 8 * hf;
            fa.u[v] = *(const uint32_t*)(pa + kb);
        }
#pragma unroll
        for (int t = 0; t < NTILES; ++t) {
            FragBF16 fb;
            const uint16_t* pb = &sB[buf][(t * 16 + r) * 32];
#pragma unroll
            for (int v = 0; v < 8; ++v) {
                int kb = ((v < 4) ? (2 * v) : (16 + 2 * (v - 4))) + 8 * hf;
                fb.u[v] = *(const uint32_t*)(pb + kb);
            }
            acc[t] = __builtin_amdgcn_wmma_f32_16x16x32_bf16(
                false, fa.v, false, fb.v, (short)0, acc[t], false, false);
        }
        __syncthreads();                              // buffer reuse fence
    }

    // C/D layout: VGPR j: lanes 0-15 -> M=j, lanes 16-31 -> M=j+8 ; N = lane%16
#pragma unroll
    for (int t = 0; t < NTILES; ++t) {
        float* pc = C + (size_t)(m0 + wv * 16 + 8 * hf) * N + (n0 + t * 16 + r);
#pragma unroll
        for (int j = 0; j < 8; ++j) pc[(size_t)j * N] = acc[t][j];
    }
}

// ---------------- depthwise 3x3 conv + bias + SiLU, scatter into 4 scan orders ----------------
__global__ void k_dwconv_silu(const float* __restrict__ xz, const float* __restrict__ cw,
                              const float* __restrict__ cb,
                              float* __restrict__ xs_f, uint16_t* __restrict__ xs_b) {
    int bp = blockIdx.x;
    int b = bp >> 10, pix = bp & 1023;
    int h = pix >> 5, w = pix & 31;
    int d = threadIdx.x;
    float acc = 0.f;
#pragma unroll
    for (int dh = -1; dh <= 1; ++dh)
#pragma unroll
        for (int dw = -1; dw <= 1; ++dw) {
            int hh = h + dh, ww = w + dw;
            if (hh < 0 || hh > 31 || ww < 0 || ww > 31) continue;
            float v = xz[((size_t)b * LSEQ + hh * 32 + ww) * (2 * DINNER) + d];
            acc += v * cw[(size_t)d * 9 + (dh + 1) * 3 + (dw + 1)];
        }
    float s = silu(acc + cb[d]);
    uint16_t sb = f2bf(s);
    int l1 = w * 32 + h;                // transpose order
    int l0 = pix, l2 = LSEQ - 1 - pix, l3 = LSEQ - 1 - l1;
    const size_t KS = (size_t)NPIX * DINNER;
    size_t base = (size_t)b * LSEQ;
    size_t i0 = (base + l0) * DINNER + d, i1 = (base + l1) * DINNER + d;
    size_t i2 = (base + l2) * DINNER + d, i3 = (base + l3) * DINNER + d;
    xs_f[0 * KS + i0] = s;  xs_b[0 * KS + i0] = sb;
    xs_f[1 * KS + i1] = s;  xs_b[1 * KS + i1] = sb;
    xs_f[2 * KS + i2] = s;  xs_b[2 * KS + i2] = sb;
    xs_f[3 * KS + i3] = s;  xs_b[3 * KS + i3] = sb;
}

// ---------------- selective scan: 32 blocks (b,k) x 256 threads (d), 16 states in VGPRs ----------------
// x_dbl layout: [k][pos][48] : cols 0..7 dt_c, 8..23 B, 24..39 C (40..47 pad)
__global__ void k_scan(const float* __restrict__ x_dbl, const float* __restrict__ xs_f,
                       const float* __restrict__ dtw, const float* __restrict__ dtb,
                       const float* __restrict__ Aneg, const float* __restrict__ Ds,
                       float* __restrict__ oy) {
    int blk = blockIdx.x;
    int b = blk >> 2, k = blk & 3;
    int d = threadIdx.x;
    float w8[DTRANK];
#pragma unroll
    for (int r = 0; r < DTRANK; ++r) w8[r] = dtw[((size_t)k * DINNER + d) * DTRANK + r];
    float bias = dtb[(size_t)k * DINNER + d];
    float Ar[DSTATE];
#pragma unroll
    for (int s = 0; s < DSTATE; ++s) Ar[s] = Aneg[((size_t)k * DINNER + d) * DSTATE + s];
    float Dv = Ds[(size_t)k * DINNER + d];
    float hS[DSTATE];
#pragma unroll
    for (int s = 0; s < DSTATE; ++s) hS[s] = 0.f;

    const float* xd  = x_dbl + ((size_t)k * NPIX + (size_t)b * LSEQ) * CPAD;
    const float* us  = xs_f  + ((size_t)k * NPIX + (size_t)b * LSEQ) * DINNER + d;
    float*       out = oy    + (((size_t)b * KDIR + k) * LSEQ) * DINNER + d;

    for (int l = 0; l < LSEQ; ++l) {
        const float* c40 = xd + (size_t)l * CPAD;   // broadcast reads (L0/L2 hit)
        float din = bias;
#pragma unroll
        for (int r = 0; r < DTRANK; ++r) din += w8[r] * c40[r];
        float delta = (din > 20.f) ? din : log1pf(__expf(din));   // softplus
        float u  = us[(size_t)l * DINNER];
        float du = delta * u;
        float y  = 0.f;
#pragma unroll
        for (int s = 0; s < DSTATE; ++s) {
            float dA = __expf(delta * Ar[s]);
            hS[s] = hS[s] * dA + du * c40[8 + s];
            y += hS[s] * c40[24 + s];
        }
        out[(size_t)l * DINNER] = y + u * Dv;
    }
}

// ---------------- merge 4 directions + LayerNorm + SiLU(z) gate -> bf16 ----------------
__global__ void k_merge_ln_gate(const float* __restrict__ oy, const float* __restrict__ xz,
                                const float* __restrict__ g, const float* __restrict__ be,
                                uint16_t* __restrict__ yg) {
    int bp = blockIdx.x;
    int b = bp >> 10, pix = bp & 1023;
    int h = pix >> 5, w = pix & 31;
    int lT = w * 32 + h;
    int d = threadIdx.x;
    size_t bb = (size_t)b * KDIR * LSEQ * DINNER;
    auto at = [&](int k, int l) { return oy[bb + ((size_t)k * LSEQ + l) * DINNER + d]; };
    float y = at(0, pix) + at(2, LSEQ - 1 - pix) + at(1, lT) + at(3, LSEQ - 1 - lT);

    __shared__ float red[DINNER];
    red[d] = y; __syncthreads();
#pragma unroll
    for (int off = 128; off > 0; off >>= 1) { if (d < off) red[d] += red[d + off]; __syncthreads(); }
    float mu = red[0] * (1.f / DINNER); __syncthreads();
    float dy = y - mu;
    red[d] = dy * dy; __syncthreads();
#pragma unroll
    for (int off = 128; off > 0; off >>= 1) { if (d < off) red[d] += red[d + off]; __syncthreads(); }
    float var = red[0] * (1.f / DINNER);
    float yn = dy * rsqrtf(var + 1e-5f) * g[d] + be[d];
    float z = xz[(size_t)bp * (2 * DINNER) + DINNER + d];
    yg[(size_t)bp * DINNER + d] = f2bf(yn * silu(z));
}

// ---------------- final epilogue: +bias, transpose (B,L,128) -> (B,128,32,32) ----------------
__global__ void k_final(const float* __restrict__ y2, const float* __restrict__ cbias,
                        float* __restrict__ out) {
    int bp = blockIdx.x;
    int b = bp >> 10, pix = bp & 1023;
    int o = threadIdx.x;   // 0..127
    out[((size_t)b * DMODEL + o) * LSEQ + pix] = y2[(size_t)bp * DMODEL + o] + cbias[o];
}

// ==================================================================================
extern "C" void kernel_launch(void* const* d_in, const int* in_sizes, int n_in,
                              void* d_out, int out_size, void* d_ws, size_t ws_size,
                              hipStream_t stream) {
    (void)in_sizes; (void)n_in; (void)out_size; (void)ws_size;
    const float* x         = (const float*)d_in[0];
    const float* proj_w    = (const float*)d_in[1];
    const float* bn_gamma  = (const float*)d_in[2];
    const float* bn_beta   = (const float*)d_in[3];
    const float* bn_mean   = (const float*)d_in[4];
    const float* bn_var    = (const float*)d_in[5];
    const float* in_proj_w = (const float*)d_in[6];
    const float* conv2d_w  = (const float*)d_in[7];
    const float* conv2d_b  = (const float*)d_in[8];
    const float* x_proj_w  = (const float*)d_in[9];
    const float* dt_projs_w= (const float*)d_in[10];
    const float* dt_projs_b= (const float*)d_in[11];
    const float* A_logs    = (const float*)d_in[12];
    const float* Ds        = (const float*)d_in[13];
    const float* ln_gamma  = (const float*)d_in[14];
    const float* ln_beta   = (const float*)d_in[15];
    const float* out_proj_w= (const float*)d_in[16];
    const float* out_conv_w= (const float*)d_in[17];
    const float* out_conv_b= (const float*)d_in[18];
    float* out = (float*)d_out;

    // ---- workspace carve-up ----
    size_t off = 0;
    auto alloc = [&](size_t bytes) -> char* {
        off = (off + 255) & ~(size_t)255;
        char* p = (char*)d_ws + off;
        off += bytes;
        return p;
    };
    uint16_t* xe      = (uint16_t*)alloc((size_t)NPIX * DMODEL * 2);        // bf16 tokens
    uint16_t* w_in    = (uint16_t*)alloc((size_t)512 * DMODEL * 2);
    uint16_t* w_xp    = (uint16_t*)alloc((size_t)KDIR * CPAD * DINNER * 2);
    uint16_t* w_op    = (uint16_t*)alloc((size_t)DMODEL * DINNER * 2);
    uint16_t* w_oc    = (uint16_t*)alloc((size_t)DMODEL * DMODEL * 2);
    float*    Aneg    = (float*)   alloc((size_t)KDIR * DINNER * DSTATE * 4);
    float*    xz      = (float*)   alloc((size_t)NPIX * 512 * 4);
    float*    xs_f    = (float*)   alloc((size_t)KDIR * NPIX * DINNER * 4);
    uint16_t* xs_b    = (uint16_t*)alloc((size_t)KDIR * NPIX * DINNER * 2);
    float*    x_dbl   = (float*)   alloc((size_t)KDIR * NPIX * CPAD * 4);
    float*    oy      = (float*)   alloc((size_t)KDIR * NPIX * DINNER * 4);
    uint16_t* yg      = (uint16_t*)alloc((size_t)NPIX * DINNER * 2);
    float*    y1      = (float*)   alloc((size_t)NPIX * DMODEL * 4);
    uint16_t* y1b     = (uint16_t*)alloc((size_t)NPIX * DMODEL * 2);
    float*    y2      = (float*)   alloc((size_t)NPIX * DMODEL * 4);

    // ---- prep: weight conversions / A = -exp(A_logs) ----
    k_f32_to_bf16<<<(512 * DMODEL + 255) / 256, 256, 0, stream>>>(in_proj_w, w_in, 512 * DMODEL);
    k_xproj_pad  <<<(KDIR * CPAD * DINNER + 255) / 256, 256, 0, stream>>>(x_proj_w, w_xp);
    k_f32_to_bf16<<<(DMODEL * DINNER + 255) / 256, 256, 0, stream>>>(out_proj_w, w_op, DMODEL * DINNER);
    k_f32_to_bf16<<<(DMODEL * DMODEL + 255) / 256, 256, 0, stream>>>(out_conv_w, w_oc, DMODEL * DMODEL);
    k_aneg       <<<(KDIR * DINNER * DSTATE + 255) / 256, 256, 0, stream>>>(A_logs, Aneg, KDIR * DINNER * DSTATE);

    // ---- patch embed + BN -> bf16 tokens ----
    k_patch_embed<<<NPIX, DMODEL, 0, stream>>>(x, proj_w, bn_gamma, bn_beta, bn_mean, bn_var, xe);

    // ---- in_proj GEMM: (8192x128)*(512x128)^T -> xz (8192x512) ----
    k_gemm_staged<8><<<dim3(NPIX / 128, 512 / 128), 256, 0, stream>>>(xe, w_in, xz, NPIX, 512, DMODEL);

    // ---- depthwise conv + SiLU -> 4-direction sequences ----
    k_dwconv_silu<<<NPIX, DINNER, 0, stream>>>(xz, conv2d_w, conv2d_b, xs_f, xs_b);

    // ---- x_proj GEMMs per direction: (8192x256)*(48x256)^T -> x_dbl_k (8192x48) ----
    for (int k = 0; k < KDIR; ++k) {
        k_gemm_staged<3><<<dim3(NPIX / 128, 1), 256, 0, stream>>>(
            xs_b + (size_t)k * NPIX * DINNER,
            w_xp + (size_t)k * CPAD * DINNER,
            x_dbl + (size_t)k * NPIX * CPAD,
            NPIX, CPAD, DINNER);
    }

    // ---- selective scan (serial over L, parallel over 8192 channels) ----
    k_scan<<<BATCH * KDIR, DINNER, 0, stream>>>(x_dbl, xs_f, dt_projs_w, dt_projs_b, Aneg, Ds, oy);

    // ---- merge directions + LayerNorm + gate ----
    k_merge_ln_gate<<<NPIX, DINNER, 0, stream>>>(oy, xz, ln_gamma, ln_beta, yg);

    // ---- out_proj GEMM: (8192x256)*(128x256)^T -> y1 ----
    k_gemm_staged<8><<<dim3(NPIX / 128, 1), 256, 0, stream>>>(yg, w_op, y1, NPIX, DMODEL, DINNER);
    k_f32_to_bf16<<<(NPIX * DMODEL + 255) / 256, 256, 0, stream>>>(y1, y1b, NPIX * DMODEL);

    // ---- out_conv GEMM: (8192x128)*(128x128)^T -> y2 ----
    k_gemm_staged<8><<<dim3(NPIX / 128, 1), 256, 0, stream>>>(y1b, w_oc, y2, NPIX, DMODEL, DMODEL);

    // ---- bias + NCHW transpose ----
    k_final<<<NPIX, DMODEL, 0, stream>>>(y2, out_conv_b, out);
}
